// BiMultiHeadAttention_79551384256897
// MI455X (gfx1250) — matrix operand order repaired and checked by reading, not compiled, not run on hardware
//
#include <hip/hip_runtime.h>
#include <hip/hip_bf16.h>

#define DEV __device__ __forceinline__

namespace {

// Problem constants (fixed by the reference).
constexpr int NB = 2, TV = 8192, TL = 256, VD = 256, LD = 768, EE = 2048, NH = 8, HD = 256;
constexpr float Q_SCALE   = 0.0625f;     // HD^-0.5 = 1/16
constexpr float CLAMP_VAL = 50000.0f;
constexpr float MASK_FILL = -9.0e15f;

typedef __attribute__((ext_vector_type(16))) __bf16 v16bf;
typedef __attribute__((ext_vector_type(8)))  float  v8f;

union FragU { v16bf v; unsigned u[8]; };

DEV unsigned short f2bf(float f) {           // f32 -> bf16 round-to-nearest-even
  unsigned u = __float_as_uint(f);
  u += 0x7FFFu + ((u >> 16) & 1u);
  return (unsigned short)(u >> 16);
}

// ---------------------------------------------------------------- converts
__global__ void cvt_f32_to_bf16(const float* __restrict__ in,
                                unsigned short* __restrict__ out, int n) {
  int i = blockIdx.x * blockDim.x + threadIdx.x;
  if (i < n) out[i] = f2bf(in[i]);
}

__global__ void mask_bias_kernel(const int* __restrict__ mask,
                                 float* __restrict__ biasf, int n) {
  int i = blockIdx.x * blockDim.x + threadIdx.x;
  if (i < n) {
    float mf = (float)mask[i];
    biasf[i] = (mf == 0.0f) ? MASK_FILL : mf;   // faithful: survivors get +1.0
  }
}

// ---------------------------------------------------------------- WMMA GEMM
// C[M,N] = op( A[M,K](bf16) x B(bf16) + bias[N] ) with batched (b,h) slicing.
// BT=false: B is row-major [K,N]. BT=true: B^T is row-major [N,K] (Q*K^T).
// Block: 256 threads (8 waves), 128x128 tile, wave tile 64(M)x32(N), K-step 32.
// Staging: A tiles (and B^T tiles) are copied global->LDS with
// GLOBAL_LOAD_ASYNC_TO_LDS_B128 (ASYNCcnt), double-buffered so the async unit
// fills buffer p^1 while the waves run WMMA out of buffer p.
// LDS row pitch = 80 bytes (32 bf16 + 16B pad): keeps every 16B async chunk
// aligned and spreads the 16-row fragment gathers across banks (gcd(20,64)=4).
template <bool BT, bool OUTBF16, bool DOCLAMP>
__global__ __launch_bounds__(256) void gemm_wmma_bf16(
    const unsigned short* __restrict__ Ag, const unsigned short* __restrict__ Bg,
    void* __restrict__ Cg, const float* __restrict__ bias, float scale,
    int K, int lda, int ldb, int ldc,
    long long sAb, long long sAh, long long sBb, long long sBh,
    long long sCb, long long sCh, int Hn) {
  __shared__ unsigned Asm[2][128][20];   // 2 x 128 rows x (32 bf16 + pad)
  __shared__ unsigned Bsm[2][128][20];   // N-major: row n holds K 0..31 (+pad)

  const int tid  = threadIdx.x;
  const int z    = blockIdx.z;
  const int bb   = z / Hn, hh = z % Hn;
  const unsigned short* Ab = Ag + (size_t)bb * sAb + (size_t)hh * sAh;
  const unsigned short* Bb = Bg + (size_t)bb * sBb + (size_t)hh * sBh;
  char* Cb = (char*)Cg + ((size_t)bb * sCb + (size_t)hh * sCh) * (OUTBF16 ? 2 : 4);

  const int mBase = blockIdx.y * 128, nBase = blockIdx.x * 128;
  const int wave = tid >> 5, lane = tid & 31, half = lane >> 4, l16 = lane & 15;
  const int mOff = (wave >> 2) * 64, nOff = (wave & 3) * 32;

  // LDS byte offsets (flat shared addresses truncate to the LDS offset).
  const unsigned aLds = (unsigned)(unsigned long long)&Asm[0][0][0];
  const unsigned bLds = (unsigned)(unsigned long long)&Bsm[0][0][0];
  constexpr unsigned BUFB = 128u * 20u * 4u;   // bytes per buffer

  // Stage one 32-wide K tile into buffer q.
  auto stage = [&](int kk2, int q) {
    {   // A: 128x32 bf16, contiguous K -> 2 async b128 copies per thread
      int r0 = tid >> 2, ck = (tid & 3) << 4;              // 16-byte chunk
      unsigned l0 = aLds + (unsigned)q * BUFB + (unsigned)r0 * 80u + (unsigned)ck;
      const char* g0 = (const char*)Ab + (((size_t)(mBase + r0) * lda + kk2) << 1) + ck;
      asm volatile("global_load_async_to_lds_b128 %0, %1, off" :: "v"(l0), "v"(g0) : "memory");
      unsigned l1 = l0 + 64u * 80u;
      const char* g1 = g0 + (((size_t)64 * lda) << 1);
      asm volatile("global_load_async_to_lds_b128 %0, %1, off" :: "v"(l1), "v"(g1) : "memory");
    }
    if constexpr (BT) {   // B^T row-major [N,K]: identical async copy
      int r0 = tid >> 2, ck = (tid & 3) << 4;
      unsigned l0 = bLds + (unsigned)q * BUFB + (unsigned)r0 * 80u + (unsigned)ck;
      const char* g0 = (const char*)Bb + (((size_t)(nBase + r0) * ldb + kk2) << 1) + ck;
      asm volatile("global_load_async_to_lds_b128 %0, %1, off" :: "v"(l0), "v"(g0) : "memory");
      unsigned l1 = l0 + 64u * 80u;
      const char* g1 = g0 + (((size_t)64 * ldb) << 1);
      asm volatile("global_load_async_to_lds_b128 %0, %1, off" :: "v"(l1), "v"(g1) : "memory");
    } else {              // B row-major [K,N]: coalesced read, transpose in LDS
      int k = tid >> 3, n0 = (tid & 7) << 4;
      const uint4* src = (const uint4*)(Bb + (size_t)(kk2 + k) * ldb + nBase + n0);
      uint4 d0 = src[0], d1 = src[1];
      unsigned tmp[8] = {d0.x, d0.y, d0.z, d0.w, d1.x, d1.y, d1.z, d1.w};
      unsigned short* bs = (unsigned short*)&Bsm[q][0][0];
      #pragma unroll
      for (int j = 0; j < 8; ++j) {
        bs[(size_t)(n0 + 2 * j) * 40 + k]     = (unsigned short)(tmp[j] & 0xFFFFu);
        bs[(size_t)(n0 + 2 * j + 1) * 40 + k] = (unsigned short)(tmp[j] >> 16);
      }
    }
  };

  v8f acc[4][2];
  #pragma unroll
  for (int i = 0; i < 4; ++i) { acc[i][0] = {}; acc[i][1] = {}; }

  stage(0, 0);
  int p = 0;
  for (int kk = 0; kk < K; kk += 32) {
    asm volatile("s_wait_asynccnt 0x0" ::: "memory");  // tile p landed in LDS
    __syncthreads();
    if (kk + 32 < K) stage(kk + 32, p ^ 1);            // prefetch next tile

    // B fragments (32x16): lane holds col n=l16, K pairs at dwords {4h..4h+3, 4h+8..4h+11}
    FragU bf0, bf1;
    const unsigned* br0 = &Bsm[p][nOff + l16][0];
    const unsigned* br1 = &Bsm[p][nOff + 16 + l16][0];
    #pragma unroll
    for (int v = 0; v < 8; ++v) { bf0.u[v] = br0[4 * half + v + (v & 4)]; bf1.u[v] = br1[4 * half + v + (v & 4)]; }

    #pragma unroll
    for (int mt = 0; mt < 4; ++mt) {
      // A fragment (16x32): lane holds row m=l16, K pairs at dwords
      // {4h..4h+3, 4h+8..4h+11} per ISA 16-bit A layout.
      FragU a;
      const unsigned* ar = &Asm[p][mOff + mt * 16 + l16][4 * half];
      a.u[0] = ar[0]; a.u[1] = ar[1]; a.u[2] = ar[2];  a.u[3] = ar[3];
      a.u[4] = ar[8]; a.u[5] = ar[9]; a.u[6] = ar[10]; a.u[7] = ar[11];
      acc[mt][0] = __builtin_amdgcn_wmma_f32_16x16x32_bf16(
          false, a.v, false, bf0.v, (short)0, acc[mt][0], false, false);
      acc[mt][1] = __builtin_amdgcn_wmma_f32_16x16x32_bf16(
          false, a.v, false, bf1.v, (short)0, acc[mt][1], false, false);
    }
    __syncthreads();   // all waves done reading buffer p before it is refilled
    p ^= 1;
  }

  // Epilogue. C layout: lane col n = l16, VGPR v holds row v + 8*half.
  #pragma unroll
  for (int mt = 0; mt < 4; ++mt) {
    #pragma unroll
    for (int nt = 0; nt < 2; ++nt) {
      int n = nBase + nOff + nt * 16 + l16;
      float bv = bias ? bias[n] : 0.0f;
      #pragma unroll
      for (int v = 0; v < 8; ++v) {
        int m = mBase + mOff + mt * 16 + 8 * half + v;
        float x = (acc[mt][nt][v] + bv) * scale;
        if (DOCLAMP) x = fminf(fmaxf(x, -CLAMP_VAL), CLAMP_VAL);
        if (OUTBF16) ((unsigned short*)Cb)[(size_t)m * ldc + n] = f2bf(x);
        else         ((float*)Cb)[(size_t)m * ldc + n] = x;
      }
    }
  }
}

// ---------------------------------------------------------------- softmaxes
// attn_v: softmax over TL (=256) with per-key bias. One block per row.
__global__ __launch_bounds__(256) void row_softmax_bias(
    const float* __restrict__ S, const float* __restrict__ biasf,
    unsigned short* __restrict__ P) {
  __shared__ float red[256];
  const size_t r = blockIdx.x;                 // 0 .. NB*NH*TV-1
  const int s = threadIdx.x;                   // TL == blockDim
  const int b = (int)(r / ((size_t)NH * TV));
  float x = S[r * TL + s] + biasf[b * TL + s];
  red[s] = x; __syncthreads();
  for (int o = 128; o > 0; o >>= 1) { if (s < o) red[s] = fmaxf(red[s], red[s + o]); __syncthreads(); }
  float mx = red[0]; __syncthreads();
  float p = __expf(x - mx);
  red[s] = p; __syncthreads();
  for (int o = 128; o > 0; o >>= 1) { if (s < o) red[s] += red[s + o]; __syncthreads(); }
  P[r * TL + s] = f2bf(p * (1.0f / red[0]));
}

// attn_l: softmax over the TV axis of S^T. Three passes with online partials.
constexpr int NCH = TV / 256;   // 32 chunks of 256 rows

__global__ __launch_bounds__(256) void col_softmax_p1(
    const float* __restrict__ S, float* __restrict__ pm, float* __restrict__ ps) {
  int bh = blockIdx.x, ch = blockIdx.y, s = threadIdx.x;
  const float* base = S + ((size_t)bh * TV + (size_t)ch * 256) * TL + s;
  float m = -3.4e38f, sum = 0.0f;
  for (int i = 0; i < 256; ++i) {
    float x = base[(size_t)i * TL];
    float mn = fmaxf(m, x);
    sum = sum * __expf(m - mn) + __expf(x - mn);
    m = mn;
  }
  pm[((size_t)bh * NCH + ch) * 256 + s] = m;
  ps[((size_t)bh * NCH + ch) * 256 + s] = sum;
}

__global__ __launch_bounds__(256) void col_softmax_p2(
    const float* __restrict__ pm, const float* __restrict__ ps,
    float* __restrict__ cm, float* __restrict__ cs) {
  int bh = blockIdx.x, s = threadIdx.x;
  float m = -3.4e38f;
  for (int c = 0; c < NCH; ++c) m = fmaxf(m, pm[((size_t)bh * NCH + c) * 256 + s]);
  float sum = 0.0f;
  for (int c = 0; c < NCH; ++c)
    sum += ps[((size_t)bh * NCH + c) * 256 + s] *
           __expf(pm[((size_t)bh * NCH + c) * 256 + s] - m);
  cm[(size_t)bh * 256 + s] = m;
  cs[(size_t)bh * 256 + s] = sum;
}

// Note: the reference clamps (x - max) at -50000 before exp; exp underflows to
// exactly 0 below ~-88 in f32, so the clamp is a numerical no-op and is elided.
__global__ __launch_bounds__(256) void col_softmax_p3(
    const float* __restrict__ S, const float* __restrict__ cm,
    const float* __restrict__ cs, unsigned short* __restrict__ Pl) {
  int bh = blockIdx.x, ch = blockIdx.y, s = threadIdx.x;
  float m = cm[(size_t)bh * 256 + s];
  float inv = 1.0f / cs[(size_t)bh * 256 + s];
  const float* base = S + ((size_t)bh * TV + (size_t)ch * 256) * TL + s;
  unsigned short* out = Pl + ((size_t)bh * TL + s) * TV + (size_t)ch * 256;
  for (int i = 0; i < 256; ++i)
    out[i] = f2bf(__expf(base[(size_t)i * TL] - m) * inv);
}

}  // namespace

// ---------------------------------------------------------------- launcher
extern "C" void kernel_launch(void* const* d_in, const int* in_sizes, int n_in,
                              void* d_out, int out_size, void* d_ws, size_t ws_size,
                              hipStream_t stream) {
  (void)in_sizes; (void)n_in; (void)out_size; (void)ws_size;
  const float* v    = (const float*)d_in[0];
  const float* l    = (const float*)d_in[1];
  const int*   mask = (const int*)d_in[2];
  const float* Wv   = (const float*)d_in[3];  const float* bv  = (const float*)d_in[4];
  const float* Wl   = (const float*)d_in[5];  const float* bl  = (const float*)d_in[6];
  const float* Wvv  = (const float*)d_in[7];  const float* bvv = (const float*)d_in[8];
  const float* Wvl  = (const float*)d_in[9];  const float* bvl = (const float*)d_in[10];
  const float* Wov  = (const float*)d_in[11]; const float* bov = (const float*)d_in[12];
  const float* Wol  = (const float*)d_in[13]; const float* bol = (const float*)d_in[14];

  char* ws = (char*)d_ws;
  size_t off = 0;
  auto alloc = [&](size_t bytes) { size_t o = off; off += (bytes + 255) & ~(size_t)255; return o; };

  const size_t nV  = (size_t)NB * TV * VD;       // v elems
  const size_t nL  = (size_t)NB * TL * LD;       // l elems
  const size_t nQ  = (size_t)NB * TV * EE;       // q / vv / Ov / Pv elems
  const size_t nK  = (size_t)NB * TL * EE;       // k / vl / Ol elems
  const size_t nS  = (size_t)NB * NH * TV * TL;  // S / Pl elems

  unsigned short* vbf   = (unsigned short*)(ws + alloc(nV * 2));
  unsigned short* lbf   = (unsigned short*)(ws + alloc(nL * 2));
  unsigned short* Wv_b  = (unsigned short*)(ws + alloc((size_t)VD * EE * 2));
  unsigned short* Wl_b  = (unsigned short*)(ws + alloc((size_t)LD * EE * 2));
  unsigned short* Wvv_b = (unsigned short*)(ws + alloc((size_t)VD * EE * 2));
  unsigned short* Wvl_b = (unsigned short*)(ws + alloc((size_t)LD * EE * 2));
  unsigned short* Wov_b = (unsigned short*)(ws + alloc((size_t)EE * VD * 2));
  unsigned short* Wol_b = (unsigned short*)(ws + alloc((size_t)EE * LD * 2));
  unsigned short* qbf   = (unsigned short*)(ws + alloc(nQ * 2));  // later reused as Pv
  unsigned short* kbf   = (unsigned short*)(ws + alloc(nK * 2));
  unsigned short* vvbf  = (unsigned short*)(ws + alloc(nQ * 2));
  unsigned short* vlbf  = (unsigned short*)(ws + alloc(nK * 2));
  unsigned short* Plbf  = (unsigned short*)(ws + alloc(nS * 2));
  float* S = (float*)(ws + alloc(nS * 4));   // 134 MB; dead after softmaxes:
  unsigned short* Ovbf = (unsigned short*)S;                    // alias (67 MB)
  unsigned short* Olbf = (unsigned short*)((char*)S + nQ * 2);  // alias (+2 MB)
  float* biasf = (float*)(ws + alloc((size_t)NB * TL * 4));
  float* pm = (float*)(ws + alloc((size_t)NB * NH * NCH * 256 * 4));
  float* ps = (float*)(ws + alloc((size_t)NB * NH * NCH * 256 * 4));
  float* cm = (float*)(ws + alloc((size_t)NB * NH * 256 * 4));
  float* cs = (float*)(ws + alloc((size_t)NB * NH * 256 * 4));

  // 1) converts (all counts divisible by 256)
  cvt_f32_to_bf16<<<(int)(nV / 256), 256, 0, stream>>>(v, vbf, (int)nV);
  cvt_f32_to_bf16<<<(int)(nL / 256), 256, 0, stream>>>(l, lbf, (int)nL);
  cvt_f32_to_bf16<<<VD * EE / 256, 256, 0, stream>>>(Wv,  Wv_b,  VD * EE);
  cvt_f32_to_bf16<<<LD * EE / 256, 256, 0, stream>>>(Wl,  Wl_b,  LD * EE);
  cvt_f32_to_bf16<<<VD * EE / 256, 256, 0, stream>>>(Wvv, Wvv_b, VD * EE);
  cvt_f32_to_bf16<<<LD * EE / 256, 256, 0, stream>>>(Wvl, Wvl_b, LD * EE);
  cvt_f32_to_bf16<<<EE * VD / 256, 256, 0, stream>>>(Wov, Wov_b, EE * VD);
  cvt_f32_to_bf16<<<EE * LD / 256, 256, 0, stream>>>(Wol, Wol_b, EE * LD);

  // 2) projections: q=(v@Wv+bv)*scale, k=l@Wl+bl, vv=v@Wvv+bvv, vl=l@Wvl+bvl
  gemm_wmma_bf16<false, true, false><<<dim3(EE / 128, (NB * TV) / 128, 1), 256, 0, stream>>>(
      vbf, Wv_b, qbf, bv, Q_SCALE, VD, VD, EE, EE, 0, 0, 0, 0, 0, 0, 1);
  gemm_wmma_bf16<false, true, false><<<dim3(EE / 128, (NB * TL) / 128, 1), 256, 0, stream>>>(
      lbf, Wl_b, kbf, bl, 1.0f, LD, LD, EE, EE, 0, 0, 0, 0, 0, 0, 1);
  gemm_wmma_bf16<false, true, false><<<dim3(EE / 128, (NB * TV) / 128, 1), 256, 0, stream>>>(
      vbf, Wvv_b, vvbf, bvv, 1.0f, VD, VD, EE, EE, 0, 0, 0, 0, 0, 0, 1);
  gemm_wmma_bf16<false, true, false><<<dim3(EE / 128, (NB * TL) / 128, 1), 256, 0, stream>>>(
      lbf, Wvl_b, vlbf, bvl, 1.0f, LD, LD, EE, EE, 0, 0, 0, 0, 0, 0, 1);

  // 3) S[b,h] = clamp(q_h @ k_h^T)  (B-transposed, f32 out)
  gemm_wmma_bf16<true, false, true><<<dim3(TL / 128, TV / 128, NB * NH), 256, 0, stream>>>(
      qbf, kbf, S, nullptr, 1.0f, HD, EE, EE, TL,
      (long long)TV * EE, HD, (long long)TL * EE, HD,
      (long long)NH * TV * TL, (long long)TV * TL, NH);

  // 4) mask bias + vision->language softmax (over TL, with bias) -> Pv (= qbf)
  mask_bias_kernel<<<(NB * TL) / 256, 256, 0, stream>>>(mask, biasf, NB * TL);
  row_softmax_bias<<<NB * NH * TV, 256, 0, stream>>>(S, biasf, qbf);

  // 5) language->vision softmax (over TV of S^T) -> Pl (transposed, bf16)
  col_softmax_p1<<<dim3(NB * NH, NCH), 256, 0, stream>>>(S, pm, ps);
  col_softmax_p2<<<NB * NH, 256, 0, stream>>>(pm, ps, cm, cs);
  col_softmax_p3<<<dim3(NB * NH, NCH), 256, 0, stream>>>(S, cm, cs, Plbf);

  // 6) attention-value products, head-merged outputs (S region now dead)
  gemm_wmma_bf16<false, true, false><<<dim3(HD / 128, TV / 128, NB * NH), 256, 0, stream>>>(
      qbf /*Pv*/, vlbf, Ovbf, nullptr, 1.0f, TL, TL, EE, EE,
      (long long)NH * TV * TL, (long long)TV * TL, (long long)TL * EE, HD,
      (long long)TV * EE, HD, NH);
  gemm_wmma_bf16<false, true, false><<<dim3(HD / 128, TL / 128, NB * NH), 256, 0, stream>>>(
      Plbf, vvbf, Olbf, nullptr, 1.0f, TV, TV, EE, EE,
      (long long)NH * TL * TV, (long long)TL * TV, (long long)TV * EE, HD,
      (long long)TL * EE, HD, NH);

  // 7) output projections, f32 directly into d_out (out_v then out_l, flat)
  float* out_v = (float*)d_out;
  float* out_l = out_v + (size_t)NB * TV * VD;
  gemm_wmma_bf16<false, false, false><<<dim3(VD / 128, (NB * TV) / 128, 1), 256, 0, stream>>>(
      Ovbf, Wov_b, out_v, bov, 1.0f, EE, EE, VD, VD, 0, 0, 0, 0, 0, 0, 1);
  gemm_wmma_bf16<false, false, false><<<dim3(LD / 128, (NB * TL) / 128, 1), 256, 0, stream>>>(
      Olbf, Wol_b, out_l, bol, 1.0f, EE, EE, LD, LD, 0, 0, 0, 0, 0, 0, 1);
}
